// ScaledDotProductAttention_43946105373006
// MI455X (gfx1250) — matrix operand order repaired and checked by reading, not compiled, not run on hardware
//
#include <hip/hip_runtime.h>

typedef __attribute__((ext_vector_type(16))) _Float16 v16h;
typedef __attribute__((ext_vector_type(8)))  float    v8f;

#define S_LEN 2048
#define DHEAD 64
#define BH    32                    // B*H = 2*16
#define WAVES_PER_BLOCK 8
#define ROWS_PER_BLOCK (16 * WAVES_PER_BLOCK)   // 128 query rows per block

// ---------------------------------------------------------------------------
// Pass 1: scores = clip(Q K^T / 8) masked -> streamed to attn buffer (raw),
//         per-lane online softmax stats, cross-lane combine ONCE at the end.
// One wave handles a 16-row Q tile against all 2048 keys.
// ---------------------------------------------------------------------------
__global__ __launch_bounds__(256)
void attn_pass1(const float* __restrict__ Q,
                const float* __restrict__ Kmat,
                const unsigned char* __restrict__ mask,
                float* __restrict__ scores,      // attn region of d_out (raw scores)
                float* __restrict__ m_ws,
                float* __restrict__ l_ws)
{
    const int lane = threadIdx.x & 31;
    const int wave = threadIdx.x >> 5;
    const int bh   = blockIdx.x;                               // 0..31
    const int q0   = blockIdx.y * ROWS_PER_BLOCK + wave * 16;  // 16-row tile base
    const int half = lane >> 4;                                // 0 or 1
    const int l16  = lane & 15;

    const long qk_base   = (long)bh * S_LEN * DHEAD;
    const long attn_base = (long)bh * S_LEN * (long)S_LEN;

    // ---- Q tile as two A fragments (16x32 f16 each), per ISA A layout:
    // lane half 0: row=l16, K {kb..kb+7, kb+16..kb+23} with kb = f*32
    // lane half 1: same row, K shifted by +8
    v16h a[2];
    {
        const float* qrow = Q + qk_base + (long)(q0 + l16) * DHEAD;
        #pragma unroll
        for (int f = 0; f < 2; ++f) {
            const int kb = f * 32 + half * 8;
            #pragma unroll
            for (int i = 0; i < 8; ++i) a[f][i]     = (_Float16)qrow[kb + i];
            #pragma unroll
            for (int i = 0; i < 8; ++i) a[f][8 + i] = (_Float16)qrow[kb + 16 + i];
        }
    }

    // Per-LANE online softmax state: each lane owns one column per tile, so
    // lane-local running (max, sum) over the 128 columns it sees is exact.
    float m[8], lsum[8];
    #pragma unroll
    for (int r = 0; r < 8; ++r) { m[r] = -3.0e38f; lsum[r] = 0.0f; }

    const float scale = 0.125f;   // 1/sqrt(64)

    for (int j = 0; j < S_LEN / 16; ++j) {
        // ---- B fragments: B = K^T (64x16). B[k][n] = K[j*16+n][k].
        // lane: N=l16, K = f*32 + half*8 + {0..7, 16..23} (ISA table order, matches A).
        const float* krow = Kmat + qk_base + (long)(j * 16 + l16) * DHEAD;
        // prefetch next K tile while this one computes
        if (j + 1 < S_LEN / 16)
            __builtin_prefetch(krow + 16 * DHEAD, 0, 1);

        v16h b[2];
        #pragma unroll
        for (int f = 0; f < 2; ++f) {
            const int kb = f * 32 + half * 8;
            #pragma unroll
            for (int i = 0; i < 16; ++i) b[f][i] = (_Float16)krow[kb + (i < 8 ? i : i + 8)];
        }

        v8f c = {};
        c = __builtin_amdgcn_wmma_f32_16x16x32_f16(false, a[0], false, b[0],
                                                   (short)0, c, false, false);
        c = __builtin_amdgcn_wmma_f32_16x16x32_f16(false, a[1], false, b[1],
                                                   (short)0, c, false, false);

        // C layout: element r -> (q = q0 + r + 8*half, k = j*16 + l16)
        const long off0 = attn_base + (long)(q0 + 8 * half) * S_LEN + j * 16 + l16;
        #pragma unroll
        for (int r = 0; r < 8; ++r) {
            const long off = off0 + (long)r * S_LEN;
            float s = c[r] * scale;
            s = fminf(fmaxf(s, -1.0e9f), 1.0e10f);
            const unsigned char mk = mask[off];
            s = mk ? -1.0e9f : s;
            // stream raw masked score (512MB stream: keep it out of L2's way)
            __builtin_nontemporal_store(s, &scores[off]);

            // lane-local online update (no cross-lane traffic in the hot loop)
            const float mnew = fmaxf(m[r], s);
            lsum[r] = lsum[r] * __expf(m[r] - mnew) + __expf(s - mnew);
            m[r] = mnew;
        }
    }

    // ---- single cross-lane combine per row (within each 16-lane half) ----
    #pragma unroll
    for (int r = 0; r < 8; ++r) {
        float mf = m[r];
        mf = fmaxf(mf, __shfl_xor(mf, 1));
        mf = fmaxf(mf, __shfl_xor(mf, 2));
        mf = fmaxf(mf, __shfl_xor(mf, 4));
        mf = fmaxf(mf, __shfl_xor(mf, 8));
        float lf = lsum[r] * __expf(m[r] - mf);
        lf += __shfl_xor(lf, 1);
        lf += __shfl_xor(lf, 2);
        lf += __shfl_xor(lf, 4);
        lf += __shfl_xor(lf, 8);
        m[r] = mf;
        lsum[r] = lf;
    }
    if (l16 == 0) {
        #pragma unroll
        for (int r = 0; r < 8; ++r) {
            const long idx = (long)bh * S_LEN + q0 + r + 8 * half;
            m_ws[idx] = m[r];
            l_ws[idx] = lsum[r];
        }
    }
}

// ---------------------------------------------------------------------------
// Pass 2: p = clamp(exp(s - m)/l, 1e-8, 1) written back over the raw scores,
//         O = P x V accumulated with WMMA, O -> result region of d_out.
// ---------------------------------------------------------------------------
__global__ __launch_bounds__(256)
void attn_pass2(const float* __restrict__ V,
                float* __restrict__ attn,        // raw scores in, probabilities out
                const float* __restrict__ m_ws,
                const float* __restrict__ l_ws,
                float* __restrict__ out)
{
    const int lane = threadIdx.x & 31;
    const int wave = threadIdx.x >> 5;
    const int bh   = blockIdx.x;
    const int q0   = blockIdx.y * ROWS_PER_BLOCK + wave * 16;
    const int half = lane >> 4;
    const int l16  = lane & 15;

    const long attn_base = (long)bh * S_LEN * (long)S_LEN;
    const long v_base    = (long)bh * S_LEN * DHEAD;

    // A fragments use row = l16 for both halves (halves differ only in K cols)
    const int   arow = q0 + l16;
    const float mrow = m_ws[(long)bh * S_LEN + arow];
    const float minv = 1.0f / l_ws[(long)bh * S_LEN + arow];
    float* prow = attn + attn_base + (long)arow * S_LEN;

    v8f acc[4];
    #pragma unroll
    for (int t = 0; t < 4; ++t) acc[t] = (v8f){};

    for (int j = 0; j < S_LEN; j += 32) {
        // ---- A fragment: P rows (16) x kv cols j..j+31, ISA A layout
        const int kb = j + half * 8;
        if (j + 32 < S_LEN) {
            __builtin_prefetch(&prow[kb + 32], 0, 1);
            __builtin_prefetch(&V[v_base + (long)(j + 32 + half * 16) * DHEAD + l16], 0, 1);
        }

        float pv[16];
        #pragma unroll
        for (int i = 0; i < 8; ++i) pv[i]     = __builtin_nontemporal_load(&prow[kb + i]);
        #pragma unroll
        for (int i = 0; i < 8; ++i) pv[8 + i] = __builtin_nontemporal_load(&prow[kb + 16 + i]);

        v16h a;
        #pragma unroll
        for (int i = 0; i < 16; ++i) {
            float p = __expf(pv[i] - mrow) * minv;
            p = fminf(fmaxf(p, 1.0e-8f), 1.0f);
            pv[i] = p;
            a[i]  = (_Float16)p;
        }
        // write final probabilities back (each element covered exactly once)
        #pragma unroll
        for (int i = 0; i < 8; ++i) __builtin_nontemporal_store(pv[i],     &prow[kb + i]);
        #pragma unroll
        for (int i = 0; i < 8; ++i) __builtin_nontemporal_store(pv[8 + i], &prow[kb + 16 + i]);

        // ---- B fragments: V rows j..j+31, four 16-wide column blocks.
        // lane: N=l16, K = half*8 + {0..7, 16..23}  -> B[k][n] = V[j+k][t*16+n]
        #pragma unroll
        for (int t = 0; t < 4; ++t) {
            v16h b;
            #pragma unroll
            for (int i = 0; i < 16; ++i)
                b[i] = (_Float16)V[v_base + (long)(j + half * 8 + (i < 8 ? i : i + 8)) * DHEAD + t * 16 + l16];
            acc[t] = __builtin_amdgcn_wmma_f32_16x16x32_f16(false, a, false, b,
                                                            (short)0, acc[t], false, false);
        }
    }

    // C layout store: (q = q0 + r + 8*half, col = t*16 + l16)
    #pragma unroll
    for (int t = 0; t < 4; ++t)
        #pragma unroll
        for (int r = 0; r < 8; ++r)
            out[((long)bh * S_LEN + q0 + r + 8 * half) * DHEAD + t * 16 + l16] = acc[t][r];
}

// ---------------------------------------------------------------------------
extern "C" void kernel_launch(void* const* d_in, const int* in_sizes, int n_in,
                              void* d_out, int out_size, void* d_ws, size_t ws_size,
                              hipStream_t stream) {
    const float*         Q    = (const float*)d_in[0];
    const float*         Km   = (const float*)d_in[1];
    const float*         V    = (const float*)d_in[2];
    const unsigned char* mask = (const unsigned char*)d_in[3];

    float* result = (float*)d_out;                                   // (B,H,S,D)
    float* attn   = result + (long)BH * S_LEN * DHEAD;               // (B,H,S,S)

    float* m_ws = (float*)d_ws;                                      // BH*S floats
    float* l_ws = m_ws + (long)BH * S_LEN;                           // BH*S floats

    dim3 grid(BH, S_LEN / ROWS_PER_BLOCK, 1);   // (32, 16)
    dim3 block(32 * WAVES_PER_BLOCK, 1, 1);     // 256 threads = 8 waves

    attn_pass1<<<grid, block, 0, stream>>>(Q, Km, mask, attn, m_ws, l_ws);
    attn_pass2<<<grid, block, 0, stream>>>(V, attn, m_ws, l_ws, result);
}
